// AttentionConv3D_1417339207700
// MI455X (gfx1250) — matrix operand order repaired and checked by reading, not compiled, not run on hardware
//
#include <hip/hip_runtime.h>
#include <hip/hip_bf16.h>

// ---------------- problem constants ----------------
#define NVOX 1728          // 12*12*12
#define NB   8
#define NJ   17
#define CPJ  8
#define NC   136
#define NLIMB 16
#define TEMP 10.0f
#define STEP (2500.0f / 11.0f)
#define LOG2E 1.44269504088896340736f

// conn table: conn[l] = (A,B)
__constant__ int d_connA[NLIMB] = {0,1,2,5,4,3,6,7,8,9,8,11,10,8,13,14};
__constant__ int d_connB[NLIMB] = {1,2,6,4,3,6,7,8,16,16,12,12,11,13,14,15};
// flattened GROUP[j0] = list of (j1, limb, e)
__constant__ int d_goff[NJ + 1] = {0,1,3,5,7,9,10,13,15,19,20,21,23,25,27,29,30,32};
__constant__ int d_gj1[32] = {1, 0,2, 1,6, 4,6, 5,3, 4, 2,3,7, 6,8, 7,16,12,13, 16, 11, 12,10, 8,11, 8,14, 13,15, 14, 8,9};
__constant__ int d_gl [32] = {0, 0,1, 1,2, 4,5, 3,4, 3, 2,5,6, 6,7, 7,8,10,13, 9, 12, 11,12, 10,11, 13,14, 14,15, 15, 8,9};
__constant__ int d_ge [32] = {1, 0,1, 0,1, 0,1, 0,1, 1, 0,0,1, 0,1, 0,1,1,1, 1, 1, 1,0, 0,0, 0,1, 0,1, 0, 0,0};

// ---------------- WMMA types ----------------
typedef __attribute__((ext_vector_type(16))) __bf16 v16bf;
typedef __attribute__((ext_vector_type(8)))  float  v8f;
typedef __attribute__((ext_vector_type(2)))  __bf16 bf16x2;
typedef __attribute__((ext_vector_type(2)))  float  f32x2;

union BF16Vec {
    v16bf v;
    unsigned int u[8];
    uint4 q[2];
};

// packed f32x2 -> bf16x2 (RNE) -> one dword; lowers to v_cvt_pk_bf16_f32
__device__ __forceinline__ unsigned int pkbf(float lo, float hi) {
    f32x2 f;
    f.x = lo; f.y = hi;
    bf16x2 h = __builtin_convertvector(f, bf16x2);
    return __builtin_bit_cast(unsigned int, h);
}
__device__ __forceinline__ unsigned short f2bf(float f) {
    __bf16 h = (__bf16)f;
    return __builtin_bit_cast(unsigned short, h);
}

// ---------------- K0: pairwise distance matrix (N x N), lives in L2 ----------------
__global__ void k_dis(float* __restrict__ dis) {
    int i = blockIdx.x * blockDim.x + threadIdx.x;
    if (i >= NVOX * NVOX) return;
    int m = i / NVOX, n = i - m * NVOX;
    int mx = m / 144, my = (m / 12) % 12, mz = m % 12;
    int nx = n / 144, ny = (n / 12) % 12, nz = n % 12;
    float dx = (float)(mx - nx) * STEP;
    float dy = (float)(my - ny) * STEP;
    float dz = (float)(mz - nz) * STEP;
    dis[i] = sqrtf(dx * dx + dy * dy + dz * dz);
}

// ---------------- K1: per-(b,j) logits -> softmax A, x_pi, weighted, bf16 RHS, shared ----------------
__global__ void k_stage1(const float* __restrict__ x, const float* __restrict__ w_pi,
                         const float* __restrict__ w_att, float* __restrict__ Aout,
                         float* __restrict__ xpi, unsigned short* __restrict__ Gt,
                         float* __restrict__ sharedv) {
    const int bj = blockIdx.x;              // 0..135
    const int b = bj / NJ, j = bj % NJ;
    const int tid = threadIdx.x;            // 0..255

    __shared__ float slog[NVOX];
    __shared__ float sred[256];

    float wa[CPJ], wp[CPJ][CPJ];
#pragma unroll
    for (int c = 0; c < CPJ; ++c) wa[c] = w_att[j * CPJ + c];
#pragma unroll
    for (int o = 0; o < CPJ; ++o)
#pragma unroll
        for (int c = 0; c < CPJ; ++c) wp[o][c] = w_pi[(j * CPJ + o) * CPJ + c];

    const float* xb = x + ((size_t)b * NC + j * CPJ) * NVOX;
    float lmax = -3.4e38f;
    for (int n = tid; n < NVOX; n += 256) {
        float xv[CPJ];
#pragma unroll
        for (int c = 0; c < CPJ; ++c) xv[c] = xb[(size_t)c * NVOX + n];
        float lg = 0.f;
#pragma unroll
        for (int c = 0; c < CPJ; ++c) lg += wa[c] * xv[c];
        slog[n] = lg;
        lmax = fmaxf(lmax, lg);
#pragma unroll
        for (int o = 0; o < CPJ; ++o) {
            float s = 0.f;
#pragma unroll
            for (int c = 0; c < CPJ; ++c) s += wp[o][c] * xv[c];
            xpi[(((size_t)b * NJ + j) * CPJ + o) * NVOX + n] = s;
        }
    }
    // max reduce
    sred[tid] = lmax; __syncthreads();
    for (int s = 128; s > 0; s >>= 1) { if (tid < s) sred[tid] = fmaxf(sred[tid], sred[tid + s]); __syncthreads(); }
    float gmax = sred[0]; __syncthreads();
    // sum of exp
    float lsum = 0.f;
    for (int n = tid; n < NVOX; n += 256) lsum += __expf(slog[n] - gmax);
    sred[tid] = lsum; __syncthreads();
    for (int s = 128; s > 0; s >>= 1) { if (tid < s) sred[tid] += sred[tid + s]; __syncthreads(); }
    float inv = 1.f / sred[0]; __syncthreads();

    float acc[CPJ];
#pragma unroll
    for (int o = 0; o < CPJ; ++o) acc[o] = 0.f;
    unsigned short* gtb = Gt + ((size_t)b * NJ + j) * 9 * NVOX;
    for (int n = tid; n < NVOX; n += 256) {
        float a = __expf(slog[n] - gmax) * inv;
        Aout[((size_t)b * NJ + j) * NVOX + n] = a;
        gtb[(size_t)8 * NVOX + n] = f2bf(a);
#pragma unroll
        for (int o = 0; o < CPJ; ++o) {
            float w = xpi[(((size_t)b * NJ + j) * CPJ + o) * NVOX + n] * a;
            gtb[(size_t)o * NVOX + n] = f2bf(w);
            acc[o] += w;
        }
    }
    // deterministic tree reduce of shared[o]
#pragma unroll
    for (int o = 0; o < CPJ; ++o) {
        sred[tid] = acc[o]; __syncthreads();
        for (int s = 128; s > 0; s >>= 1) { if (tid < s) sred[tid] += sred[tid + s]; __syncthreads(); }
        if (tid == 0) sharedv[((size_t)b * NJ + j) * CPJ + o] = sred[0];
        __syncthreads();
    }
}

// ---------------- K2: Econst[b][j0][o] = w_conv . shared  minus replaced-joint part ----------------
__global__ void k_econst(const float* __restrict__ w_conv, const float* __restrict__ sharedv,
                         float* __restrict__ econst) {
    int i = blockIdx.x * blockDim.x + threadIdx.x;
    if (i >= NB * NJ * CPJ) return;
    int o = i % CPJ;
    int j0 = (i / CPJ) % NJ;
    int b = i / (CPJ * NJ);
    const float* wc = w_conv + ((size_t)j0 * CPJ + o) * NC;
    const float* sh = sharedv + (size_t)b * NC;
    float s = 0.f;
    for (int c = 0; c < NC; ++c) s += wc[c] * sh[c];
    for (int k = d_goff[j0]; k < d_goff[j0 + 1]; ++k) {
        int j1 = d_gj1[k];
        for (int cc = 0; cc < CPJ; ++cc) s -= wc[j1 * CPJ + cc] * sh[j1 * CPJ + cc];
    }
    econst[i] = s;
}

// ---------------- K3: fused exp + WMMA GEMM -> feat = (P@wj)/(P@Aj) ----------------
// one wave per (limb, 16-row m-tile). 144 RHS cols = 9 tiles of 16.
// K-loop unrolled by 2 with ping-pong B buffers so a full chunk of 18 b128
// loads stays in flight behind the 9 WMMAs of the previous chunk.
__global__ void __launch_bounds__(32) k_masks(const float* __restrict__ dis,
                                              const unsigned short* __restrict__ Gt,
                                              const float* __restrict__ limb_mean,
                                              const float* __restrict__ limb_std,
                                              float* __restrict__ feat) {
    const int l    = blockIdx.y;            // limb
    const int m0   = blockIdx.x * 16;       // output-row tile
    const int lane = threadIdx.x;
    const int half = lane >> 4;
    const int lr   = lane & 15;

    const float mu = limb_mean[l];
    const float sd = limb_std[l];
    // P = exp2(qa*d^2 + qb*d + qc), qa = -log2(e)/(T*2*sd^2+0.1), folded quadratic
    const float qa = -LOG2E / (TEMP * 2.0f * sd * sd + 0.1f);
    const float qb = -2.0f * mu * qa;
    const float qc = mu * mu * qa;

    // per-lane column decode for the 9 B tiles: col = 16t + lr -> (e, b, c')
    const unsigned short* rowp[9];
    int ecol[9], bcol[9], ccol[9];
#pragma unroll
    for (int t = 0; t < 9; ++t) {
        int col = t * 16 + lr;
        int e = (col >= 72) ? 1 : 0;
        int rem = col - e * 72;
        int bb = rem / 9;
        int cc = rem - bb * 9;
        int jj = e ? d_connB[l] : d_connA[l];
        ecol[t] = e; bcol[t] = bb; ccol[t] = cc;
        rowp[t] = Gt + (((size_t)bb * NJ + jj) * 9 + cc) * (size_t)NVOX + half * 8;
    }

    const float* disrow = dis + (size_t)(m0 + lr) * NVOX; // A-operand row per lane
    const int kb1 = half * 8;        // A layout: lanes 0-15 -> K 0..7 / 16..23 ; lanes 16-31 -> K 8..15 / 24..31
    const int kb2 = 16 + half * 8;

    v8f acc[9];
#pragma unroll
    for (int t = 0; t < 9; ++t) acc[t] = (v8f){0.f, 0.f, 0.f, 0.f, 0.f, 0.f, 0.f, 0.f};

#define PEXP(d) __builtin_amdgcn_exp2f(__builtin_fmaf((d), __builtin_fmaf((d), qa, qb), qc))

    // build A operand (16x32 bf16 tile of P) for K-chunk at k
    auto makeA = [&](int k) -> v16bf {
        const float4* d1 = (const float4*)(disrow + k + kb1);
        const float4* d2 = (const float4*)(disrow + k + kb2);
        float4 da = d1[0], db = d1[1];
        float4 dc = d2[0], dd = d2[1];
        BF16Vec av;
        av.u[0] = pkbf(PEXP(da.x), PEXP(da.y));
        av.u[1] = pkbf(PEXP(da.z), PEXP(da.w));
        av.u[2] = pkbf(PEXP(db.x), PEXP(db.y));
        av.u[3] = pkbf(PEXP(db.z), PEXP(db.w));
        av.u[4] = pkbf(PEXP(dc.x), PEXP(dc.y));
        av.u[5] = pkbf(PEXP(dc.z), PEXP(dc.w));
        av.u[6] = pkbf(PEXP(dd.x), PEXP(dd.y));
        av.u[7] = pkbf(PEXP(dd.z), PEXP(dd.w));
        return av.v;
    };
    // load the 9 B tiles (32x16 bf16 each) for K-chunk at k
    auto loadB = [&](BF16Vec* bv, int k) {
#pragma unroll
        for (int t = 0; t < 9; ++t) {
            const uint4* g = (const uint4*)(rowp[t] + k);
            bv[t].q[0] = g[0];
            bv[t].q[1] = g[2];
        }
    };

    BF16Vec b0[9], b1[9];
    loadB(b0, 0);
    for (int k0 = 0; k0 < NVOX; k0 += 64) {          // 27 iterations
        loadB(b1, k0 + 32);                          // prefetch next chunk
        v16bf a0 = makeA(k0);
#pragma unroll
        for (int t = 0; t < 9; ++t)
            acc[t] = __builtin_amdgcn_wmma_f32_16x16x32_bf16(
                false, a0, false, b0[t].v, (short)0, acc[t], false, false);

        int kn = (k0 + 64 < NVOX) ? (k0 + 64) : 0;   // clamped; last-iter values unused
        loadB(b0, kn);                               // prefetch next-next chunk
        v16bf a1 = makeA(k0 + 32);
#pragma unroll
        for (int t = 0; t < 9; ++t)
            acc[t] = __builtin_amdgcn_wmma_f32_16x16x32_bf16(
                false, a1, false, b1[t].v, (short)0, acc[t], false, false);
    }
#undef PEXP

    // broadcast den (c'==8 columns) through 1KB LDS, then divide + store feat
    __shared__ float lds_den[2][8][16];
#pragma unroll
    for (int t = 0; t < 9; ++t) {
        if (ccol[t] == 8) {
#pragma unroll
            for (int r = 0; r < 8; ++r)
                lds_den[ecol[t]][bcol[t]][half * 8 + r] = acc[t][r];
        }
    }
    __syncthreads();
#pragma unroll
    for (int t = 0; t < 9; ++t) {
        if (ccol[t] < 8) {
            float* fp = feat + ((((size_t)l * 2 + ecol[t]) * NB + bcol[t]) * CPJ + ccol[t]) * (size_t)NVOX
                        + m0 + half * 8;
#pragma unroll
            for (int r = 0; r < 8; ++r)
                fp[r] = acc[t][r] / lds_den[ecol[t]][bcol[t]][half * 8 + r];
        }
    }
}

// ---------------- K4: out = Econst + x_pi + sum_edges w_conv . feat ----------------
__global__ void k_final(const float* __restrict__ w_conv, const float* __restrict__ econst,
                        const float* __restrict__ xpi, const float* __restrict__ feat,
                        float* __restrict__ out) {
    const int bj = blockIdx.x;
    const int b = bj / NJ, j0 = bj % NJ;
    const int tid = threadIdx.x;
    const int g0 = d_goff[j0], g1 = d_goff[j0 + 1];

    for (int n = tid; n < NVOX; n += 256) {
        float fv[4][CPJ];
        for (int k = g0; k < g1; ++k) {
            const float* fp = feat + (((size_t)d_gl[k] * 2 + d_ge[k]) * NB + b) * CPJ * (size_t)NVOX + n;
#pragma unroll
            for (int cc = 0; cc < CPJ; ++cc) fv[k - g0][cc] = fp[(size_t)cc * NVOX];
        }
#pragma unroll
        for (int o = 0; o < CPJ; ++o) {
            float a = econst[((size_t)b * NJ + j0) * CPJ + o]
                    + xpi[(((size_t)b * NJ + j0) * CPJ + o) * (size_t)NVOX + n];
            for (int k = g0; k < g1; ++k) {
                const float* wc = w_conv + ((size_t)j0 * CPJ + o) * NC + d_gj1[k] * CPJ;
#pragma unroll
                for (int cc = 0; cc < CPJ; ++cc) a += wc[cc] * fv[k - g0][cc];
            }
            out[((size_t)b * NC + j0 * CPJ + o) * (size_t)NVOX + n] = a;
        }
    }
}

// ---------------- workspace layout (bytes, 256-aligned) ----------------
#define OFF_DIS    ((size_t)0)                                    // 1728*1728*4  = 11,943,936
#define OFF_GT     ((size_t)11943936)                             // 8*17*9*1728*2 = 4,230,144
#define OFF_XPI    ((size_t)16174080)                             // 8*17*8*1728*4 = 7,520,256
#define OFF_FEAT   ((size_t)23694336)                             // 16*2*8*8*1728*4 = 14,155,776
#define OFF_SHARED ((size_t)37850112)                             // 8*17*8*4 = 4,352
#define OFF_ECONST ((size_t)37854464)                             // 4,352

extern "C" void kernel_launch(void* const* d_in, const int* in_sizes, int n_in,
                              void* d_out, int out_size, void* d_ws, size_t ws_size,
                              hipStream_t stream) {
    (void)in_sizes; (void)n_in; (void)out_size; (void)ws_size;
    const float* x         = (const float*)d_in[0];
    const float* w_pi      = (const float*)d_in[1];
    const float* w_att     = (const float*)d_in[2];
    const float* w_conv    = (const float*)d_in[3];
    const float* limb_mean = (const float*)d_in[4];
    const float* limb_std  = (const float*)d_in[5];

    float* out  = (float*)d_out;
    float* Aout = out + (size_t)NB * NC * NVOX;     // second tuple element

    char* ws = (char*)d_ws;
    float*          dis     = (float*)(ws + OFF_DIS);
    unsigned short* Gt      = (unsigned short*)(ws + OFF_GT);
    float*          xpi     = (float*)(ws + OFF_XPI);
    float*          feat    = (float*)(ws + OFF_FEAT);
    float*          sharedv = (float*)(ws + OFF_SHARED);
    float*          econst  = (float*)(ws + OFF_ECONST);

    k_dis<<<(NVOX * NVOX + 255) / 256, 256, 0, stream>>>(dis);
    k_stage1<<<NB * NJ, 256, 0, stream>>>(x, w_pi, w_att, Aout, xpi, Gt, sharedv);
    k_econst<<<(NB * NJ * CPJ + 63) / 64, 64, 0, stream>>>(w_conv, sharedv, econst);
    dim3 gm(NVOX / 16, NLIMB);
    k_masks<<<gm, 32, 0, stream>>>(dis, Gt, limb_mean, limb_std, feat);
    k_final<<<NB * NJ, 256, 0, stream>>>(w_conv, econst, xpi, feat, out);
}
